// IntegratedJODIELayer_32770600468498
// MI455X (gfx1250) — matrix operand order repaired and checked by reading, not compiled, not run on hardware
//
#include <hip/hip_runtime.h>
#include <hip/hip_bf16.h>

typedef _Float16 v16h __attribute__((ext_vector_type(16)));
typedef _Float16 v4h  __attribute__((ext_vector_type(4)));
typedef float    v8f  __attribute__((ext_vector_type(8)));
typedef unsigned v8u  __attribute__((ext_vector_type(8)));

#define JB      16384
#define JNODES  1000000
#define MEMD    128
#define TIMED   100

// ---------------------------------------------------------------------------
// Weight packing: f32 KxN (N=128) -> f16 WMMA B-fragment order.
// Fragment (kt, nt) = 32x16 tile. For lane L, element e (0..15):
//   K = kt*32 + ((L>>4)<<3) + (e&7) + ((e>>3)<<4) ,  N = nt*16 + (L&15)
// Linear layout: dst[((kt*8 + nt)*32 + lane)*16 + e]
// swap_half: source k-tile = kt^4 for kt<8 (involution) -- lets the item-side
// layer-1 GEMM read the X tile sequentially with no index swizzle in the loop.
// ---------------------------------------------------------------------------
__global__ void jodie_pack(const float* __restrict__ W, _Float16* __restrict__ dst,
                           int Korig, int Kpad, int swap_half) {
    int idx = blockIdx.x * blockDim.x + threadIdx.x;
    int total = Kpad * 128;
    if (idx >= total) return;
    int e    = idx & 15;
    int lane = (idx >> 4) & 31;
    int frag = idx >> 9;
    int nt   = frag & 7;
    int kt   = frag >> 3;
    int kt_src = (swap_half && kt < 8) ? (kt ^ 4) : kt;
    int n = nt * 16 + (lane & 15);
    int k = kt_src * 32 + ((lane >> 4) << 3) + (e & 7) + ((e >> 3) << 4);
    float v = (k < Korig) ? W[(long)k * 128 + n] : 0.0f;
    dst[idx] = (_Float16)v;
}

// ---------------------------------------------------------------------------
// A fragment: 16x32 f16 tile from row-major LDS buffer (ld elements).
// Dense 16-bit A layout (ISA 7.12.2): lane m=L&15, khalf=(L>=16)*8,
//   vgpr j: K = kt*32 + (j>>2)*16 + (j&3)*2 + khalf  (pair K, K+1)
// ---------------------------------------------------------------------------
__device__ __forceinline__ v16h load_a_frag(const _Float16* lds, int ld, int kt, int lane) {
    int m  = lane & 15;
    int kb = kt * 32 + ((lane >> 4) << 3);
    const _Float16* row = lds + m * ld;
    v8u au;
#pragma unroll
    for (int j = 0; j < 8; ++j) {
        int k = kb + ((j >> 2) << 4) + ((j & 3) << 1);
        au[j] = *(const unsigned*)(row + k);   // pair (K, K+1), 4B aligned
    }
    return __builtin_bit_cast(v16h, au);
}

// 16-row x 64-col slab per wave: acc[nt] (nt=0..3) over nkt (even) k-tiles.
// Ping-pong double buffering of A (LDS) and B (global) fragments: each buffer
// is fully consumed before refill, so no register copies are needed and the
// WMMA burst for tile kt only waits on loads issued one tile earlier.
__device__ __forceinline__ void run_gemm(const _Float16* Alds, int lda, int nkt2,
                                         const v16h* Bp, v8f acc[4], int lane,
                                         int ntbase) {
#pragma unroll
    for (int nt = 0; nt < 4; ++nt) {
        v8f z = {0.f, 0.f, 0.f, 0.f, 0.f, 0.f, 0.f, 0.f};
        acc[nt] = z;
    }
    const v16h* Bl = Bp + (size_t)ntbase * 32 + lane;  // frag (kt,nt) at Bl[kt*256 + nt*32]

    v16h a0 = load_a_frag(Alds, lda, 0, lane);
    v16h b0[4], b1[4];
#pragma unroll
    for (int nt = 0; nt < 4; ++nt) b0[nt] = Bl[nt * 32];

    for (int kt2 = 0; kt2 < nkt2; ++kt2) {
        int kt = kt2 * 2;
        // prefetch odd tile
        v16h a1 = load_a_frag(Alds, lda, kt + 1, lane);
#pragma unroll
        for (int nt = 0; nt < 4; ++nt) b1[nt] = Bl[(kt + 1) * 256 + nt * 32];
        // consume even tile
#pragma unroll
        for (int nt = 0; nt < 4; ++nt)
            acc[nt] = __builtin_amdgcn_wmma_f32_16x16x32_f16(
                false, a0, false, b0[nt], (short)0, acc[nt], false, false);
        // prefetch next even tile
        if (kt2 + 1 < nkt2) {
            a0 = load_a_frag(Alds, lda, kt + 2, lane);
#pragma unroll
            for (int nt = 0; nt < 4; ++nt) b0[nt] = Bl[(kt + 2) * 256 + nt * 32];
        }
        // consume odd tile
#pragma unroll
        for (int nt = 0; nt < 4; ++nt)
            acc[nt] = __builtin_amdgcn_wmma_f32_16x16x32_f16(
                false, a1, false, b1[nt], (short)0, acc[nt], false, false);
    }
}

// C/D layout: VGPR r -> row = r + ((L>=16)<<3), col = nt*16 + (L&15)
__device__ __forceinline__ void store_acc_f16(const v8f acc[4], _Float16* out, int ldo,
                                              const float* bias, bool relu,
                                              int lane, int ntbase) {
    int n0 = lane & 15;
    int rbase = (lane >> 4) << 3;
#pragma unroll
    for (int nt = 0; nt < 4; ++nt) {
        int col = (ntbase + nt) * 16 + n0;
        float bv = bias[col];
#pragma unroll
        for (int r = 0; r < 8; ++r) {
            float v = acc[nt][r] + bv;
            if (relu) v = fmaxf(v, 0.0f);
            out[(rbase + r) * ldo + col] = (_Float16)v;
        }
    }
}

// ---------------------------------------------------------------------------
// Main fused kernel: 2 waves per 16-row tile (each wave owns 4 n-tiles),
// full 6-GEMM chain via WMMA with LDS-resident activations.
// ---------------------------------------------------------------------------
__global__ __launch_bounds__(64) void jodie_main(
    const float* __restrict__ src_emb, const float* __restrict__ dst_emb,
    const int* __restrict__ src_ids,   const int* __restrict__ dst_ids,
    const float* __restrict__ edge,    const float* __restrict__ ts,
    const float* __restrict__ memory,  const float* __restrict__ tw,
    const float* __restrict__ tb,
    const float* __restrict__ bu1, const float* __restrict__ bu2,
    const float* __restrict__ bi1, const float* __restrict__ bi2,
    const float* __restrict__ bp1, const float* __restrict__ bp2,
    const float* __restrict__ bt1, const float* __restrict__ bo,
    const v16h* __restrict__ Wu1p, const v16h* __restrict__ Wi1p,
    const v16h* __restrict__ Wu2p, const v16h* __restrict__ Wi2p,
    const v16h* __restrict__ Wp1p, const v16h* __restrict__ Wp2p,
    const v16h* __restrict__ Wt1p, const v16h* __restrict__ Wop,
    float* __restrict__ out, float* __restrict__ user_t_ws, float* __restrict__ item_t_ws)
{
    __shared__ _Float16 X [16 * 512];   // [src_mem | dst_mem | edge | time | 0pad]
    __shared__ _Float16 Y [16 * 256];   // 228/256-wide staging
    __shared__ _Float16 Z0[16 * 128];
    __shared__ _Float16 Z1[16 * 128];
    __shared__ int   sids[32];          // [src x16 | dst x16]
    __shared__ float sts [16];

    const int tid    = threadIdx.x;
    const int lane   = tid & 31;
    const int ntbase = (tid >> 5) * 4;  // wave 0 -> n-tiles 0..3, wave 1 -> 4..7
    const int r0     = blockIdx.x * 16;

    if (tid < 16)       sids[tid]      = src_ids[r0 + tid];
    else if (tid < 32)  sids[tid]      = dst_ids[r0 + tid - 16];
    else if (tid < 48)  sts [tid - 32] = ts[r0 + tid - 32];
    __syncthreads();

    // Build concat input tile (f16), zero-padded to K=512; float4 granularity.
    for (int i4 = tid * 4; i4 < 16 * 512; i4 += 64 * 4) {
        int row = i4 >> 9, col = i4 & 511;
        int g = r0 + row;
        float4 v;
        if (col < 128) {
            v = *(const float4*)(memory + (long)sids[row] * MEMD + col);
        } else if (col < 256) {
            v = *(const float4*)(memory + (long)sids[16 + row] * MEMD + (col - 128));
        } else if (col < 384) {
            v = *(const float4*)(edge + (long)g * MEMD + (col - 256));
        } else if (col < 384 + TIMED) {   // [384, 484): 4-aligned region
            int c = col - 384;
            float t = sts[row];
            v.x = __builtin_cosf(t * tw[c + 0] + tb[c + 0]);
            v.y = __builtin_cosf(t * tw[c + 1] + tb[c + 1]);
            v.z = __builtin_cosf(t * tw[c + 2] + tb[c + 2]);
            v.w = __builtin_cosf(t * tw[c + 3] + tb[c + 3]);
        } else {
            v = make_float4(0.f, 0.f, 0.f, 0.f);
        }
        v4h h = { (_Float16)v.x, (_Float16)v.y, (_Float16)v.z, (_Float16)v.w };
        *(v4h*)(X + i4) = h;
    }
    __syncthreads();

    for (int side = 0; side < 2; ++side) {
        const v16h* W1p = side ? Wi1p : Wu1p;   // Wi1p pre-permuted: no k swizzle needed
        const v16h* W2p = side ? Wi2p : Wu2p;
        const float* b1 = side ? bi1 : bu1;
        const float* b2 = side ? bi2 : bu2;
        const float* emb = side ? dst_emb : src_emb;
        float* tws  = side ? item_t_ws : user_t_ws;
        float* outp = out + (side ? (long)JB * MEMD : 0);

        v8f acc[4];

        // layer1: relu(concat @ W1 + b1), K=512 (16 k-tiles)
        run_gemm(X, 512, 8, W1p, acc, lane, ntbase);
        store_acc_f16(acc, Z0, 128, b1, true, lane, ntbase);
        __syncthreads();

        // layer2: upd = Z0 @ W2 + b2, K=128 (4 k-tiles)
        run_gemm(Z0, 128, 2, W2p, acc, lane, ntbase);
        __syncthreads();
        store_acc_f16(acc, Z1, 128, b2, false, lane, ntbase);
        __syncthreads();

        // proj1: relu(upd @ Wp1 + bp1)
        run_gemm(Z1, 128, 2, Wp1p, acc, lane, ntbase);
        __syncthreads();
        store_acc_f16(acc, Z0, 128, bp1, true, lane, ntbase);
        __syncthreads();

        // proj2: proj = Z0 @ Wp2 + bp2 -> Y[:,0:128]
        run_gemm(Z0, 128, 2, Wp2p, acc, lane, ntbase);
        store_acc_f16(acc, Y, 256, bp2, false, lane, ntbase);

        // Y[:,128:256] = [time_emb | 0pad]
        for (int i = tid; i < 16 * 128; i += 64) {
            int row = i >> 7, c = i & 127;
            _Float16 v = (_Float16)0.0f;
            if (c < TIMED) v = X[row * 512 + 384 + c];
            Y[row * 256 + 128 + c] = v;
        }
        __syncthreads();

        // t-batch: t = relu(Y @ Wt1 + bt1), K=256 (8 k-tiles, 228 real)
        run_gemm(Y, 256, 4, Wt1p, acc, lane, ntbase);
        __syncthreads();
        {   // f32 -> workspace (for scatter) and f16 -> Y[:,0:128] (for Wo)
            int n0 = lane & 15, rbase = (lane >> 4) << 3;
#pragma unroll
            for (int nt = 0; nt < 4; ++nt) {
                int col = (ntbase + nt) * 16 + n0;
                float bv = bt1[col];
#pragma unroll
                for (int r = 0; r < 8; ++r) {
                    int row = rbase + r;
                    float v = fmaxf(acc[nt][r] + bv, 0.0f);
                    tws[(long)(r0 + row) * MEMD + col] = v;
                    Y[row * 256 + col] = (_Float16)v;
                }
            }
        }
        // Y[:,128:256] = node embedding
        for (int i = tid; i < 16 * 128; i += 64) {
            int row = i >> 7, c = i & 127;
            Y[row * 256 + 128 + c] = (_Float16)emb[(long)(r0 + row) * MEMD + c];
        }
        __syncthreads();

        // output proj: [t | emb] @ Wo + bo, K=256 (8 k-tiles) -> f32 out
        run_gemm(Y, 256, 4, Wop, acc, lane, ntbase);
        {
            int n0 = lane & 15, rbase = (lane >> 4) << 3;
#pragma unroll
            for (int nt = 0; nt < 4; ++nt) {
                int col = (ntbase + nt) * 16 + n0;
                float bv = bo[col];
#pragma unroll
                for (int r = 0; r < 8; ++r)
                    outp[(long)(r0 + rbase + r) * MEMD + col] = acc[nt][r] + bv;
            }
        }
        __syncthreads();
    }
}

// ---------------------------------------------------------------------------
// Deterministic scatter: winner[id] = max priority (side*B + row); dst > src.
// ---------------------------------------------------------------------------
__global__ void jodie_mark(const int* __restrict__ src_ids, const int* __restrict__ dst_ids,
                           int* __restrict__ winner) {
    int i = blockIdx.x * blockDim.x + threadIdx.x;
    if (i >= 2 * JB) return;
    int id = (i < JB) ? src_ids[i] : dst_ids[i - JB];
    atomicMax(&winner[id], i);
}

__global__ void jodie_scatter(const int* __restrict__ src_ids, const int* __restrict__ dst_ids,
                              const int* __restrict__ winner,
                              const float* __restrict__ user_t, const float* __restrict__ item_t,
                              float* __restrict__ mem) {
    long i = (long)blockIdx.x * blockDim.x + threadIdx.x;
    if (i >= (long)2 * JB * MEMD) return;
    int rowp = (int)(i >> 7);
    int c    = (int)(i & 127);
    int side = rowp >= JB;
    int row  = side ? rowp - JB : rowp;
    int id   = side ? dst_ids[row] : src_ids[row];
    if (winner[id] == rowp)
        mem[(long)id * MEMD + c] = (side ? item_t : user_t)[(long)row * MEMD + c];
}

// ---------------------------------------------------------------------------
extern "C" void kernel_launch(void* const* d_in, const int* in_sizes, int n_in,
                              void* d_out, int out_size, void* d_ws, size_t ws_size,
                              hipStream_t stream) {
    const float* src_emb = (const float*)d_in[0];
    const float* dst_emb = (const float*)d_in[1];
    const int*   src_ids = (const int*)d_in[2];
    const int*   dst_ids = (const int*)d_in[3];
    const float* edge    = (const float*)d_in[4];
    const float* ts      = (const float*)d_in[5];
    const float* memory  = (const float*)d_in[6];
    const float* tw      = (const float*)d_in[7];
    const float* tb      = (const float*)d_in[8];
    const float* Wu1 = (const float*)d_in[9];   const float* bu1 = (const float*)d_in[10];
    const float* Wu2 = (const float*)d_in[11];  const float* bu2 = (const float*)d_in[12];
    const float* Wi1 = (const float*)d_in[13];  const float* bi1 = (const float*)d_in[14];
    const float* Wi2 = (const float*)d_in[15];  const float* bi2 = (const float*)d_in[16];
    const float* Wp1 = (const float*)d_in[17];  const float* bp1 = (const float*)d_in[18];
    const float* Wp2 = (const float*)d_in[19];  const float* bp2 = (const float*)d_in[20];
    const float* Wt1 = (const float*)d_in[21];  const float* bt1 = (const float*)d_in[22];
    const float* Wo  = (const float*)d_in[23];  const float* bo  = (const float*)d_in[24];

    float* out     = (float*)d_out;
    float* new_mem = out + (size_t)2 * JB * MEMD;

    // Workspace layout
    char* ws = (char*)d_ws;
    _Float16* pw = (_Float16*)ws;                       // 262144 f16 = 512 KB packed weights
    float* user_t = (float*)(ws + (512 << 10));         // B*128 f32
    float* item_t = user_t + (size_t)JB * MEMD;         // B*128 f32
    int*   winner = (int*)(ws + (512 << 10) + (size_t)2 * JB * MEMD * sizeof(float));

    // 1) new_memory starts as a copy of memory (512 MB, D2D, capture-safe)
    hipMemcpyAsync(new_mem, memory, (size_t)JNODES * MEMD * sizeof(float),
                   hipMemcpyDeviceToDevice, stream);
    // 2) winner init to -1
    hipMemsetAsync(winner, 0xFF, (size_t)JNODES * sizeof(int), stream);

    // 3) pack weights into WMMA B-fragment f16 layout (K padded to mult. of 32)
    //    Wi1 packed with its first 8 k-tiles swapped (0..3 <-> 4..7).
    struct PackDesc { const float* W; int Ko; int Kp; size_t off; int swap; };
    const PackDesc packs[8] = {
        { Wu1, 484, 512,      0, 0 }, { Wi1, 484, 512,  65536, 1 },
        { Wu2, 128, 128, 131072, 0 }, { Wi2, 128, 128, 147456, 0 },
        { Wp1, 128, 128, 163840, 0 }, { Wp2, 128, 128, 180224, 0 },
        { Wt1, 228, 256, 196608, 0 }, { Wo,  256, 256, 229376, 0 },
    };
    for (int p = 0; p < 8; ++p) {
        int total = packs[p].Kp * 128;
        jodie_pack<<<(total + 255) / 256, 256, 0, stream>>>(
            packs[p].W, pw + packs[p].off, packs[p].Ko, packs[p].Kp, packs[p].swap);
    }

    // 4) fused WMMA chain: 1024 blocks x 2 waves, 16 rows / 2 n-slabs each
    jodie_main<<<JB / 16, 64, 0, stream>>>(
        src_emb, dst_emb, src_ids, dst_ids, edge, ts, memory, tw, tb,
        bu1, bu2, bi1, bi2, bp1, bp2, bt1, bo,
        (const v16h*)(pw +      0), (const v16h*)(pw +  65536),
        (const v16h*)(pw + 131072), (const v16h*)(pw + 147456),
        (const v16h*)(pw + 163840), (const v16h*)(pw + 180224),
        (const v16h*)(pw + 196608), (const v16h*)(pw + 229376),
        out, user_t, item_t);

    // 5) deterministic scatter into new_memory
    jodie_mark<<<(2 * JB + 255) / 256, 256, 0, stream>>>(src_ids, dst_ids, winner);
    long nscatter = (long)2 * JB * MEMD;
    jodie_scatter<<<(int)((nscatter + 255) / 256), 256, 0, stream>>>(
        src_ids, dst_ids, winner, user_t, item_t, new_mem);
}
